// MultiScaleBlock_14353780703403
// MI455X (gfx1250) — matrix operand order repaired and checked by reading, not compiled, not run on hardware
//
#include <hip/hip_runtime.h>
#include <cstdint>

// Problem constants (from reference)
#define NN   20000
#define EE   256000
#define EF   276000      // EE + NN (self loops appended)
#define ESM  1280
#define HH   4
#define CC   256
#define HC   1024

// ---------------------------------------------------------------------------
// bf16 helpers (stored as raw ushort; split-precision: v = hi + lo, both bf16)
// ---------------------------------------------------------------------------
typedef __attribute__((ext_vector_type(8)))  __bf16 bfx8;
typedef __attribute__((ext_vector_type(16))) __bf16 bfx16;
typedef __attribute__((ext_vector_type(8)))  float  fx8;

static __device__ __forceinline__ unsigned short f2bf(float f) {
  unsigned u = __float_as_uint(f);
  u += 0x7FFFu + ((u >> 16) & 1u);          // round-to-nearest-even
  return (unsigned short)(u >> 16);
}
static __device__ __forceinline__ float bf2f(unsigned short b) {
  return __uint_as_float(((unsigned)b) << 16);
}

// order-preserving float<->uint encoding for atomicMax-based segment max
static __device__ __forceinline__ unsigned fenc(float f) {
  unsigned u = __float_as_uint(f);
  return (u & 0x80000000u) ? ~u : (u | 0x80000000u);
}
static __device__ __forceinline__ float fdec(unsigned e) {
  unsigned u = (e & 0x80000000u) ? (e & 0x7FFFFFFFu) : ~e;
  return __uint_as_float(u);
}

// ---------------------------------------------------------------------------
// WMMA fragment loads (contiguous k order chosen here, same for A and B, wave32)
// A 16x32 bf16, row-major source (ld = K): lane m=lane&15, half=lane>>4;
//   elems 0..15 = A[m][k + 16*half .. +16)
// B 32x16 bf16, N-major (pre-transposed) source (ld = K): lane n=lane&15;
//   elems 0..15 = Bt[n][k + 16*half .. +16)
// ---------------------------------------------------------------------------
static __device__ __forceinline__ bfx16 cat8(bfx8 a, bfx8 b) {
  return __builtin_shufflevector(a, b, 0,1,2,3,4,5,6,7,8,9,10,11,12,13,14,15);
}
static __device__ __forceinline__ bfx16 load_afrag(const unsigned short* row, int k, int half) {
  bfx8 a = *(const bfx8*)(row + k + 16*half);
  bfx8 b = *(const bfx8*)(row + k + 16*half + 8);
  return cat8(a, b);
}
static __device__ __forceinline__ bfx16 load_bfrag(const unsigned short* col, int k, int half) {
  bfx8 a = *(const bfx8*)(col + k + 16*half);
  bfx8 b = *(const bfx8*)(col + k + 16*half + 8);
  return cat8(a, b);
}
static __device__ __forceinline__ fx8 wmma_bf16(bfx16 a, bfx16 b, fx8 c) {
  return __builtin_amdgcn_wmma_f32_16x16x32_bf16(false, a, false, b, (short)0, c, false, false);
}

// ---------------------------------------------------------------------------
// Elementwise split / transpose-split kernels
// ---------------------------------------------------------------------------
__global__ void k_split(const float* __restrict__ in,
                        unsigned short* __restrict__ hi,
                        unsigned short* __restrict__ lo, int n) {
  int t = blockIdx.x * 256 + threadIdx.x;
  if (t >= n) return;
  float v = in[t];
  unsigned short h = f2bf(v);
  hi[t] = h;
  lo[t] = f2bf(v - bf2f(h));
}

// in: (R x Cn) row-major  ->  out: (Cn x R), i.e. N-major weights for B frags
__global__ void k_splitT(const float* __restrict__ in,
                         unsigned short* __restrict__ hi,
                         unsigned short* __restrict__ lo, int R, int Cn) {
  int t = blockIdx.x * 256 + threadIdx.x;
  if (t >= R * Cn) return;
  int r = t / Cn, c = t % Cn;
  float v = in[t];
  unsigned short h = f2bf(v);
  size_t o = (size_t)c * R + r;
  hi[o] = h;
  lo[o] = f2bf(v - bf2f(h));
}

// ---------------------------------------------------------------------------
// Degree / self-loop attr
// ---------------------------------------------------------------------------
__global__ void k_deg(const int* __restrict__ ei, const float* __restrict__ ew,
                      float* __restrict__ deg, float* __restrict__ sumew) {
  int e = blockIdx.x * 256 + threadIdx.x;
  if (e >= EE) return;
  int d = ei[EE + e];
  atomicAdd(&deg[d], 1.f);
  atomicAdd(&sumew[d], ew[e]);
}
__global__ void k_loopattr(float* __restrict__ sumew, const float* __restrict__ deg) {
  int i = blockIdx.x * 256 + threadIdx.x;
  if (i >= NN) return;
  sumew[i] = sumew[i] / fmaxf(deg[i], 1.f);
}
__global__ void k_amax_init(unsigned* __restrict__ a) {
  int i = blockIdx.x * 256 + threadIdx.x;
  if (i < NN * HH) a[i] = 0x007FFFFFu;     // fenc(-inf)
}

// ---------------------------------------------------------------------------
// GEMM1: out1 = x@Wl + bl ; out2 = x@Wr  (M=NN, K=ESM, N=HC), shared A frags.
// split-precision: acc += Ahi*Bhi + Alo*Bhi + Ahi*Blo
// block = 8 waves -> 16 x 128 output tile; grid = (NN/16, HC/128)
// ---------------------------------------------------------------------------
__global__ void __launch_bounds__(256) k_gemm1(
    const unsigned short* __restrict__ Ah, const unsigned short* __restrict__ Al,
    const unsigned short* __restrict__ B1h, const unsigned short* __restrict__ B1l,
    const unsigned short* __restrict__ B2h, const unsigned short* __restrict__ B2l,
    const float* __restrict__ bias1,
    float* __restrict__ out1, float* __restrict__ out2) {
  const int lane = threadIdx.x & 31;
  const int wave = threadIdx.x >> 5;
  const int half = lane >> 4;
  const int l16  = lane & 15;
  const int m0 = blockIdx.x * 16;
  const int n0 = blockIdx.y * 128 + wave * 16;
  const unsigned short* arh = Ah  + (size_t)(m0 + l16) * ESM;
  const unsigned short* arl = Al  + (size_t)(m0 + l16) * ESM;
  const unsigned short* p1h = B1h + (size_t)(n0 + l16) * ESM;
  const unsigned short* p1l = B1l + (size_t)(n0 + l16) * ESM;
  const unsigned short* p2h = B2h + (size_t)(n0 + l16) * ESM;
  const unsigned short* p2l = B2l + (size_t)(n0 + l16) * ESM;
  fx8 acc1 = {0.f,0.f,0.f,0.f,0.f,0.f,0.f,0.f};
  fx8 acc2 = {0.f,0.f,0.f,0.f,0.f,0.f,0.f,0.f};
  for (int k = 0; k < ESM; k += 32) {
    bfx16 ah = load_afrag(arh, k, half);
    bfx16 al = load_afrag(arl, k, half);
    bfx16 bh = load_bfrag(p1h, k, half);
    bfx16 bl = load_bfrag(p1l, k, half);
    acc1 = wmma_bf16(ah, bh, acc1);
    acc1 = wmma_bf16(al, bh, acc1);
    acc1 = wmma_bf16(ah, bl, acc1);
    bh = load_bfrag(p2h, k, half);
    bl = load_bfrag(p2l, k, half);
    acc2 = wmma_bf16(ah, bh, acc2);
    acc2 = wmma_bf16(al, bh, acc2);
    acc2 = wmma_bf16(ah, bl, acc2);
  }
  const float bz = bias1[n0 + l16];
#pragma unroll
  for (int r = 0; r < 8; ++r) {
    const int m = m0 + r + 8 * half;       // C/D layout: lanes16-31 hold M=8..15
    out1[(size_t)m * HC + n0 + l16] = acc1[r] + bz;
    out2[(size_t)m * HC + n0 + l16] = acc2[r];
  }
}

// ---------------------------------------------------------------------------
// Edge phase: wave per edge (8 waves / block)
// ---------------------------------------------------------------------------
__global__ void k_logits(const int* __restrict__ ei, const float* __restrict__ ew,
                         const float* __restrict__ loopw,
                         const float* __restrict__ xl, const float* __restrict__ xr,
                         const float* __restrict__ We, const float* __restrict__ att,
                         float* __restrict__ logits, unsigned* __restrict__ amax) {
  const int lane = threadIdx.x & 31;
  const int f = blockIdx.x * 8 + (threadIdx.x >> 5);
  if (f >= EF) return;
  int s, d; float ea;
  if (f < EE) { s = ei[f]; d = ei[EE + f]; ea = ew[f]; }
  else        { s = f - EE; d = s; ea = loopw[s]; }
  const float* pl = xl + (size_t)s * HC;
  const float* pr = xr + (size_t)d * HC;
  for (int h = 0; h < HH; ++h) {
    float a = 0.f;
    for (int c = lane; c < CC; c += 32) {
      int j = h * CC + c;
      float m = pl[j] + pr[j] + ea * We[j];
      m = (m > 0.f) ? m : 0.2f * m;        // leaky_relu(0.2)
      a += m * att[j];
    }
    for (int o = 16; o > 0; o >>= 1) a += __shfl_xor(a, o, 32);
    if (lane == 0) {
      logits[(size_t)f * HH + h] = a;
      atomicMax(&amax[(size_t)d * HH + h], fenc(a));
    }
  }
}

__global__ void k_exp_den(const int* __restrict__ ei, float* __restrict__ logits,
                          const unsigned* __restrict__ amax, float* __restrict__ den) {
  int t = blockIdx.x * 256 + threadIdx.x;
  if (t >= EF * HH) return;
  int f = t >> 2, h = t & 3;
  int d = (f < EE) ? ei[EE + f] : (f - EE);
  float ex = expf(logits[t] - fdec(amax[(size_t)d * HH + h]));
  logits[t] = ex;                           // reuse buffer as exp(logit - amax)
  atomicAdd(&den[(size_t)d * HH + h], ex);
}

__global__ void k_gat(const int* __restrict__ ei, const float* __restrict__ exl,
                      const float* __restrict__ den, const float* __restrict__ xl,
                      float* __restrict__ gat) {
  const int lane = threadIdx.x & 31;
  const int f = blockIdx.x * 8 + (threadIdx.x >> 5);
  if (f >= EF) return;
  int s, d;
  if (f < EE) { s = ei[f]; d = ei[EE + f]; } else { s = f - EE; d = s; }
  const float* pl = xl + (size_t)s * HC;
  float* pg = gat + (size_t)d * HC;
  for (int h = 0; h < HH; ++h) {
    float al = exl[(size_t)f * HH + h] / (den[(size_t)d * HH + h] + 1e-16f);
    for (int c = lane; c < CC; c += 32) {
      int j = h * CC + c;
      atomicAdd(&pg[j], al * pl[j]);
    }
  }
}

__global__ void k_elu(float* __restrict__ gat, const float* __restrict__ bg) {
  int t = blockIdx.x * 256 + threadIdx.x;   // exactly NN*HC threads
  float g = gat[t] + bg[t & (HC - 1)];
  gat[t] = (g > 0.f) ? g : (expf(g) - 1.f);
}

__global__ void k_agg(const int* __restrict__ ei, const float* __restrict__ h,
                      float* __restrict__ agg) {
  const int lane = threadIdx.x & 31;
  const int e = blockIdx.x * 8 + (threadIdx.x >> 5);
  if (e >= EE) return;
  const float* ph = h + (size_t)ei[e] * HC;
  float* pa = agg + (size_t)ei[EE + e] * HC;
  for (int j = lane; j < HC; j += 32) atomicAdd(&pa[j], ph[j]);
}

// agg/deg normalize + bf16-split both h and agg for GEMM2
__global__ void k_finalize(const float* __restrict__ deg, const float* __restrict__ agg,
                           const float* __restrict__ h,
                           unsigned short* __restrict__ agg_hi, unsigned short* __restrict__ agg_lo,
                           unsigned short* __restrict__ h_hi,   unsigned short* __restrict__ h_lo) {
  int t = blockIdx.x * 256 + threadIdx.x;   // exactly NN*HC threads
  float dm = fmaxf(deg[t >> 10], 1.f);
  float av = agg[t] / dm;
  unsigned short ah = f2bf(av);
  agg_hi[t] = ah; agg_lo[t] = f2bf(av - bf2f(ah));
  float hv = h[t];
  unsigned short hh = f2bf(hv);
  h_hi[t] = hh; h_lo[t] = f2bf(hv - bf2f(hh));
}

// ---------------------------------------------------------------------------
// GEMM2: out = relu(agg@Wsl + h@Wsr + bsl)  (M=NN, K=HC, N=ESM)
// ---------------------------------------------------------------------------
__global__ void __launch_bounds__(256) k_gemm2(
    const unsigned short* __restrict__ A1h, const unsigned short* __restrict__ A1l,
    const unsigned short* __restrict__ A2h, const unsigned short* __restrict__ A2l,
    const unsigned short* __restrict__ B1h, const unsigned short* __restrict__ B1l,
    const unsigned short* __restrict__ B2h, const unsigned short* __restrict__ B2l,
    const float* __restrict__ bias, float* __restrict__ out) {
  const int lane = threadIdx.x & 31;
  const int wave = threadIdx.x >> 5;
  const int half = lane >> 4;
  const int l16  = lane & 15;
  const int m0 = blockIdx.x * 16;
  const int n0 = blockIdx.y * 128 + wave * 16;
  const unsigned short* a1h = A1h + (size_t)(m0 + l16) * HC;
  const unsigned short* a1l = A1l + (size_t)(m0 + l16) * HC;
  const unsigned short* a2h = A2h + (size_t)(m0 + l16) * HC;
  const unsigned short* a2l = A2l + (size_t)(m0 + l16) * HC;
  const unsigned short* b1h = B1h + (size_t)(n0 + l16) * HC;
  const unsigned short* b1l = B1l + (size_t)(n0 + l16) * HC;
  const unsigned short* b2h = B2h + (size_t)(n0 + l16) * HC;
  const unsigned short* b2l = B2l + (size_t)(n0 + l16) * HC;
  fx8 acc = {0.f,0.f,0.f,0.f,0.f,0.f,0.f,0.f};
  for (int k = 0; k < HC; k += 32) {
    bfx16 xh = load_afrag(a1h, k, half);
    bfx16 xl = load_afrag(a1l, k, half);
    bfx16 wh = load_bfrag(b1h, k, half);
    bfx16 wl = load_bfrag(b1l, k, half);
    acc = wmma_bf16(xh, wh, acc);
    acc = wmma_bf16(xl, wh, acc);
    acc = wmma_bf16(xh, wl, acc);
    xh = load_afrag(a2h, k, half);
    xl = load_afrag(a2l, k, half);
    wh = load_bfrag(b2h, k, half);
    wl = load_bfrag(b2l, k, half);
    acc = wmma_bf16(xh, wh, acc);
    acc = wmma_bf16(xl, wh, acc);
    acc = wmma_bf16(xh, wl, acc);
  }
  const float bz = bias[n0 + l16];
#pragma unroll
  for (int r = 0; r < 8; ++r) {
    const int m = m0 + r + 8 * half;
    float v = acc[r] + bz;
    out[(size_t)m * ESM + n0 + l16] = fmaxf(v, 0.f);   // relu
  }
}

// ---------------------------------------------------------------------------
// residual + LayerNorm over last dim (block per row, 1280 = 256*5)
// ---------------------------------------------------------------------------
__global__ void __launch_bounds__(256) k_layernorm(
    const float* __restrict__ x, const float* __restrict__ gamma,
    const float* __restrict__ beta, float* __restrict__ out) {
  const int row = blockIdx.x;
  const int tid = threadIdx.x;
  float z[5];
  float s = 0.f, s2 = 0.f;
#pragma unroll
  for (int j = 0; j < 5; ++j) {
    int c = tid + 256 * j;
    float v = out[(size_t)row * ESM + c] + x[(size_t)row * ESM + c];
    z[j] = v; s += v; s2 += v * v;
  }
  __shared__ float r1[256], r2[256];
  r1[tid] = s; r2[tid] = s2; __syncthreads();
  for (int o = 128; o > 0; o >>= 1) {
    if (tid < o) { r1[tid] += r1[tid + o]; r2[tid] += r2[tid + o]; }
    __syncthreads();
  }
  const float mu  = r1[0] * (1.f / ESM);
  const float var = r2[0] * (1.f / ESM) - mu * mu;
  const float inv = rsqrtf(var + 1e-5f);
#pragma unroll
  for (int j = 0; j < 5; ++j) {
    int c = tid + 256 * j;
    out[(size_t)row * ESM + c] = (z[j] - mu) * inv * gamma[c] + beta[c];
  }
}

// ---------------------------------------------------------------------------
// host launch (graph-capture safe: only async ops on `stream`)
// workspace requirement: ~375 MB (with aliasing, see layout below)
// ---------------------------------------------------------------------------
extern "C" void kernel_launch(void* const* d_in, const int* in_sizes, int n_in,
                              void* d_out, int out_size, void* d_ws, size_t ws_size,
                              hipStream_t stream) {
  (void)in_sizes; (void)n_in; (void)out_size; (void)ws_size;
  const float* x    = (const float*)d_in[0];
  const int*   ei   = (const int*)  d_in[1];
  const float* ew   = (const float*)d_in[2];
  const float* Wl   = (const float*)d_in[3];
  const float* bl   = (const float*)d_in[4];
  const float* Wr   = (const float*)d_in[5];
  const float* We   = (const float*)d_in[6];
  const float* att  = (const float*)d_in[7];
  const float* bg   = (const float*)d_in[8];
  const float* Wsl  = (const float*)d_in[9];
  const float* bsl  = (const float*)d_in[10];
  const float* Wsr  = (const float*)d_in[11];
  const float* gmm  = (const float*)d_in[12];
  const float* bta  = (const float*)d_in[13];
  float* out = (float*)d_out;

  char* base = (char*)d_ws;
  size_t off = 0;
  auto alloc = [&](size_t bytes) -> char* {
    char* p = base + off;
    off += (bytes + 255) & ~(size_t)255;
    return p;
  };
  float*    deg    = (float*)   alloc((size_t)NN * 4);
  float*    loopw  = (float*)   alloc((size_t)NN * 4);        // sum(ew) then loop_attr
  unsigned* amax   = (unsigned*)alloc((size_t)NN * HH * 4);
  float*    den    = (float*)   alloc((size_t)NN * HH * 4);
  float*    logits = (float*)   alloc((size_t)EF * HH * 4);   // logits then exp
  float*    xl     = (float*)   alloc((size_t)NN * HC * 4);   // later aliased: agg (fp32)
  float*    xr     = (float*)   alloc((size_t)NN * HC * 4);   // later aliased: agg_hi|agg_lo
  float*    gat    = (float*)   alloc((size_t)NN * HC * 4);   // becomes h after ELU
  unsigned short* xhi  = (unsigned short*)alloc((size_t)NN * ESM * 2); // later h_hi
  unsigned short* xlo  = (unsigned short*)alloc((size_t)NN * ESM * 2); // later h_lo
  unsigned short* wlth = (unsigned short*)alloc((size_t)ESM * HC * 2);
  unsigned short* wltl = (unsigned short*)alloc((size_t)ESM * HC * 2);
  unsigned short* wrth = (unsigned short*)alloc((size_t)ESM * HC * 2);
  unsigned short* wrtl = (unsigned short*)alloc((size_t)ESM * HC * 2);
  unsigned short* wslth= (unsigned short*)alloc((size_t)HC * ESM * 2);
  unsigned short* wsltl= (unsigned short*)alloc((size_t)HC * ESM * 2);
  unsigned short* wsrth= (unsigned short*)alloc((size_t)HC * ESM * 2);
  unsigned short* wsrtl= (unsigned short*)alloc((size_t)HC * ESM * 2);
  // aliases (last use of xl = k_gat, of xr = k_logits, of xhi/xlo = k_gemm1)
  float* agg = xl;
  unsigned short* agg_hi = (unsigned short*)xr;
  unsigned short* agg_lo = agg_hi + (size_t)NN * HC;
  unsigned short* hhi = xhi;
  unsigned short* hlo = xlo;

  // zero accumulators (stream-ordered, capture-safe)
  hipMemsetAsync(deg,   0, (size_t)NN * 4, stream);
  hipMemsetAsync(loopw, 0, (size_t)NN * 4, stream);
  hipMemsetAsync(den,   0, (size_t)NN * HH * 4, stream);
  hipMemsetAsync(gat,   0, (size_t)NN * HC * 4, stream);
  k_amax_init<<<(NN * HH + 255) / 256, 256, 0, stream>>>(amax);

  // degree / self-loop attr
  k_deg     <<<EE / 256, 256, 0, stream>>>(ei, ew, deg, loopw);
  k_loopattr<<<(NN + 255) / 256, 256, 0, stream>>>(loopw, deg);

  // split-precision bf16 operands
  k_split <<<(NN * ESM) / 256, 256, 0, stream>>>(x, xhi, xlo, NN * ESM);
  k_splitT<<<(ESM * HC + 255) / 256, 256, 0, stream>>>(Wl,  wlth,  wltl,  ESM, HC);
  k_splitT<<<(ESM * HC + 255) / 256, 256, 0, stream>>>(Wr,  wrth,  wrtl,  ESM, HC);
  k_splitT<<<(HC * ESM + 255) / 256, 256, 0, stream>>>(Wsl, wslth, wsltl, HC, ESM);
  k_splitT<<<(HC * ESM + 255) / 256, 256, 0, stream>>>(Wsr, wsrth, wsrtl, HC, ESM);

  // xl = x@Wl + bl ; xr = x@Wr     (WMMA, shared A frags)
  k_gemm1<<<dim3(NN / 16, HC / 128), 256, 0, stream>>>(
      xhi, xlo, wlth, wltl, wrth, wrtl, bl, xl, xr);

  // attention edge phase
  k_logits <<<EF / 8, 256, 0, stream>>>(ei, ew, loopw, xl, xr, We, att, logits, amax);
  k_exp_den<<<(EF * HH + 255) / 256, 256, 0, stream>>>(ei, logits, amax, den);
  k_gat    <<<EF / 8, 256, 0, stream>>>(ei, logits, den, xl, gat);
  k_elu    <<<(NN * HC) / 256, 256, 0, stream>>>(gat, bg);

  // neighbor mean (agg aliases xl; safe: xl's last use was k_gat)
  hipMemsetAsync(agg, 0, (size_t)NN * HC * 4, stream);
  k_agg     <<<EE / 8, 256, 0, stream>>>(ei, gat, agg);
  k_finalize<<<(NN * HC) / 256, 256, 0, stream>>>(deg, agg, gat, agg_hi, agg_lo, hhi, hlo);

  // y = relu(agg@Wsl + h@Wsr + bsl)  (WMMA) -> d_out, then residual+LN in place
  k_gemm2<<<dim3(NN / 16, ESM / 128), 256, 0, stream>>>(
      agg_hi, agg_lo, hhi, hlo, wslth, wsltl, wsrth, wsrtl, bsl, out);
  k_layernorm<<<NN, 256, 0, stream>>>(x, gmm, bta, out);
}
